// VoxelJAFAR_15599321219359
// MI455X (gfx1250) — matrix operand order repaired and checked by reading, not compiled, not run on hardware
//
#include <hip/hip_runtime.h>

#define MPTS 100000
#define KNBR 27
#define DDIM 64
#define CSEM 96
#define NCLS 13

typedef __attribute__((ext_vector_type(16))) _Float16 v16h;
typedef __attribute__((ext_vector_type(8)))  _Float16 h8;
typedef __attribute__((ext_vector_type(4)))  _Float16 h4;
typedef __attribute__((ext_vector_type(2)))  _Float16 h2;
typedef __attribute__((ext_vector_type(8)))  float    v8f;

// ---------- WMMA fragment helpers (CDNA5 wave32, 16x16x32 f16) ----------

__device__ __forceinline__ v8f zero8() {
  v8f z;
#pragma unroll
  for (int i = 0; i < 8; ++i) z[i] = 0.0f;
  return z;
}

// A 16x32 f16 tile fragment: lane L -> row = L&15; per lane two contiguous
// 8-half runs: k = koff..koff+7 and koff+16..koff+23, koff = ks*32 + (L>>4)*8.
// Tile must be row-major f16 with 16B-aligned rows (ld*2 % 16 == 0).
__device__ __forceinline__ v16h load_a_frag(const _Float16* tile, int ld, int ks, int lane) {
  const int row  = lane & 15;
  const int koff = ks * 32 + ((lane >> 4) << 3);
  const _Float16* p = tile + row * ld + koff;
  const h8 lo = *(const h8*)(p);
  const h8 hi = *(const h8*)(p + 16);
  return __builtin_shufflevector(lo, hi, 0, 1, 2, 3, 4, 5, 6, 7,
                                         8, 9, 10, 11, 12, 13, 14, 15);
}

// B 32x16 f16 fragment from an N-major staged weight (w[n*ldk + k]):
// lane L -> n = nt*16 + (L&15); two 8-half runs: k = koff..koff+7 and koff+16..koff+23, koff = ks*32 + (L>>4)*8.
__device__ __forceinline__ v16h load_b_frag(const _Float16* wnm, int ldk, int nt, int ks, int lane) {
  const int n    = (nt << 4) + (lane & 15);
  const int koff = ks * 32 + ((lane >> 4) << 3);
  const _Float16* p = wnm + n * ldk + koff;
  const h8 lo = *(const h8*)(p);
  const h8 hi = *(const h8*)(p + 16);
  return __builtin_shufflevector(lo, hi, 0, 1, 2, 3, 4, 5, 6, 7,
                                         8, 9, 10, 11, 12, 13, 14, 15);
}

// C/D 16x16 f32: VGPR r, lanes 0-15 -> M=r, N=lane; lanes 16-31 -> M=8+r.
__device__ __forceinline__ void store_c_f32(float* dst, int ld, int nt, int lane, const v8f c) {
  const int rbase = (lane >> 4) << 3;
  const int col   = (nt << 4) + (lane & 15);
#pragma unroll
  for (int r = 0; r < 8; ++r) dst[(rbase + r) * ld + col] = c[r];
}

__device__ __forceinline__ v8f wmma_f16(v16h a, v16h b, v8f c) {
  return __builtin_amdgcn_wmma_f32_16x16x32_f16(false, a, false, b, (short)0, c, false, false);
}

// ---------- Kernel 1: Qg = relu(LN(geo@W_geo)), bdy, Qp, q2=Qp@Wk^T, pe=Qp@posT ----------

__global__ __launch_bounds__(128) void k1_geo(
    const float* __restrict__ geo,  const float* __restrict__ Wgeo,
    const float* __restrict__ gma,  const float* __restrict__ bta,
    const float* __restrict__ Wbdy, const float* __restrict__ bbdy,
    const float* __restrict__ Wq,   const float* __restrict__ Wk,
    const float* __restrict__ pos_emb,
    float* __restrict__ Qg, float* __restrict__ q2, float* __restrict__ pe,
    float* __restrict__ out_bdy)
{
  __shared__ __align__(16) _Float16 sWgeoT[64 * 72];  // N-major: [n][k]
  __shared__ __align__(16) _Float16 sWqT[64 * 72];
  __shared__ __align__(16) _Float16 sWkNM[64 * 72];   // B=Wk^T, N-major == Wk rows
  __shared__ __align__(16) _Float16 sPeNM[32 * 72];
  __shared__ __align__(16) float    sF32[4][16 * 66];
  __shared__ __align__(16) _Float16 sF16[4][16 * 72];

  const int tid = threadIdx.x;
  const int lane = tid & 31;
  const int w = tid >> 5;

  for (int i = tid; i < 64 * 64; i += 128) {
    const int a = i >> 6, b = i & 63;
    // Wgeo/Wq: B[k][n] = W[k*64+n]  -> N-major slot [n][k] with k=a, n=b
    sWgeoT[b * 72 + a] = (_Float16)Wgeo[i];
    sWqT[b * 72 + a]   = (_Float16)Wq[i];
    // Wk^T:  B[k][n] = Wk[n*64+k]  -> N-major slot [n][k] with n=a, k=b
    sWkNM[a * 72 + b]  = (_Float16)Wk[i];
  }
  for (int i = tid; i < 32 * 64; i += 128) {
    const int n = i >> 6, k = i & 63;
    sPeNM[n * 72 + k] = (n < KNBR) ? (_Float16)pos_emb[n * 64 + k] : (_Float16)0.0f;
  }

  int t = blockIdx.x * 4 + w;
  if (t > MPTS / 16 - 1) t = MPTS / 16 - 1;          // clamp: uniform barriers
  const int m0 = t * 16;

  _Float16* a16 = sF16[w];
  float*    a32 = sF32[w];

  // stage geo tile: float4 loads -> h4 (b64) LDS stores
  for (int idx = lane; idx < 16 * 16; idx += 32) {
    const int r = idx >> 4, c4 = (idx & 15) << 2;
    const float4 g = *(const float4*)(geo + (size_t)(m0 + r) * 64 + c4);
    h4 y; y[0] = (_Float16)g.x; y[1] = (_Float16)g.y;
    y[2] = (_Float16)g.z; y[3] = (_Float16)g.w;
    *(h4*)(a16 + r * 72 + c4) = y;
  }
  __syncthreads();

  // GEMM1: geo_tile @ W_geo
  v8f acc[4];
#pragma unroll
  for (int nt = 0; nt < 4; ++nt) acc[nt] = zero8();
#pragma unroll
  for (int ks = 0; ks < 2; ++ks) {
    v16h a = load_a_frag(a16, 72, ks, lane);
#pragma unroll
    for (int nt = 0; nt < 4; ++nt)
      acc[nt] = wmma_f16(a, load_b_frag(sWgeoT, 72, nt, ks, lane), acc[nt]);
  }
#pragma unroll
  for (int nt = 0; nt < 4; ++nt) store_c_f32(a32, 66, nt, lane, acc[nt]);
  __syncthreads();

  // LayerNorm + ReLU in place (2 lanes per row), + boundary head
  {
    const int row = lane & 15, half = lane >> 4;
    float* rp = &a32[row * 66 + half * 32];
    float s = 0.0f, sq = 0.0f;
#pragma unroll
    for (int j = 0; j < 32; ++j) { const float x = rp[j]; s += x; sq += x * x; }
    s  += __shfl_xor(s, 16, 32);
    sq += __shfl_xor(sq, 16, 32);
    const float mu  = s * (1.0f / 64.0f);
    const float inv = rsqrtf(sq * (1.0f / 64.0f) - mu * mu + 1e-5f);
    float bdy = 0.0f;
#pragma unroll
    for (int j = 0; j < 32; ++j) {
      const int col = half * 32 + j;
      float v = (rp[j] - mu) * inv * gma[col] + bta[col];
      v = fmaxf(v, 0.0f);
      rp[j] = v;
      bdy += v * Wbdy[col];
    }
    bdy += __shfl_xor(bdy, 16, 32);
    if (half == 0) out_bdy[m0 + row] = bdy + bbdy[0];
  }
  __syncthreads();

  // coalesced Qg store + f16 tile refresh
  for (int idx = lane; idx < 16 * 32; idx += 32) {
    const int r = idx >> 5, c2 = (idx & 31) << 1;
    const float2 x = *(const float2*)(a32 + r * 66 + c2);
    *(float2*)(Qg + (size_t)(m0 + r) * 64 + c2) = x;
    h2 y; y[0] = (_Float16)x.x; y[1] = (_Float16)x.y;
    *(h2*)(a16 + r * 72 + c2) = y;
  }
  __syncthreads();

  // GEMM2: Qp = Qg_tile @ W_q
#pragma unroll
  for (int nt = 0; nt < 4; ++nt) acc[nt] = zero8();
#pragma unroll
  for (int ks = 0; ks < 2; ++ks) {
    v16h a = load_a_frag(a16, 72, ks, lane);
#pragma unroll
    for (int nt = 0; nt < 4; ++nt)
      acc[nt] = wmma_f16(a, load_b_frag(sWqT, 72, nt, ks, lane), acc[nt]);
  }
#pragma unroll
  for (int nt = 0; nt < 4; ++nt) store_c_f32(a32, 66, nt, lane, acc[nt]);
  __syncthreads();
  for (int idx = lane; idx < 16 * 32; idx += 32) {
    const int r = idx >> 5, c2 = (idx & 31) << 1;
    const float2 x = *(const float2*)(a32 + r * 66 + c2);
    h2 y; y[0] = (_Float16)x.x; y[1] = (_Float16)x.y;
    *(h2*)(a16 + r * 72 + c2) = y;
  }
  __syncthreads();

  const v16h aq0 = load_a_frag(a16, 72, 0, lane);
  const v16h aq1 = load_a_frag(a16, 72, 1, lane);

  // q2 = Qp @ Wk^T  -> ws
#pragma unroll
  for (int nt = 0; nt < 4; ++nt) {
    v8f c = zero8();
    c = wmma_f16(aq0, load_b_frag(sWkNM, 72, nt, 0, lane), c);
    c = wmma_f16(aq1, load_b_frag(sWkNM, 72, nt, 1, lane), c);
    const int rb = (lane >> 4) << 3, col = (nt << 4) + (lane & 15);
#pragma unroll
    for (int r = 0; r < 8; ++r)
      q2[(size_t)(m0 + rb + r) * 64 + col] = c[r];
  }
  // pe = Qp @ pos_emb^T (N padded to 32) -> ws
#pragma unroll
  for (int nt = 0; nt < 2; ++nt) {
    v8f c = zero8();
    c = wmma_f16(aq0, load_b_frag(sPeNM, 72, nt, 0, lane), c);
    c = wmma_f16(aq1, load_b_frag(sPeNM, 72, nt, 1, lane), c);
    const int rb = (lane >> 4) << 3, col = (nt << 4) + (lane & 15);
#pragma unroll
    for (int r = 0; r < 8; ++r)
      pe[(size_t)(m0 + rb + r) * 32 + col] = c[r];
  }
}

// ---------- Kernel 2: attention logits, softmax, sem aggregation (wave per point) ----------

__global__ __launch_bounds__(128) void k2_attn(
    const float* __restrict__ Qg, const float* __restrict__ q2,
    const float* __restrict__ pe, const float* __restrict__ sem,
    const int* __restrict__ coords, const long long* __restrict__ nbr,
    float* __restrict__ out_aff, float* __restrict__ out_nidx,
    float* __restrict__ out_vmask, float* __restrict__ vsum)
{
  const int lane = threadIdx.x & 31;
  const int m = blockIdx.x * 4 + (threadIdx.x >> 5);
  if (m >= MPTS) return;

  const float2 q2v = ((const float2*)(q2 + (size_t)m * 64))[lane];
  const int cx = coords[m * 3 + 0], cy = coords[m * 3 + 1], cz = coords[m * 3 + 2];

  int nidx = m, valid = 1, posi = 13;
  if (lane < KNBR) {
    nidx = (int)nbr[(size_t)m * KNBR + lane];
    int rx = coords[nidx * 3 + 0] - cx;
    int ry = coords[nidx * 3 + 1] - cy;
    int rz = coords[nidx * 3 + 2] - cz;
    const int ax = rx < 0 ? -rx : rx, ay = ry < 0 ? -ry : ry, az = rz < 0 ? -rz : rz;
    valid = (ax <= 1) && (ay <= 1) && (az <= 1);
    rx = min(max(rx + 1, 0), 2); ry = min(max(ry + 1, 0), 2); rz = min(max(rz + 1, 0), 2);
    posi = rx * 9 + ry * 3 + rz;
  }

  // logit_k = Qg[n_k] . q2[m]  (coalesced 256B row broadcast-gather per k)
  float logit = -1e30f;
#pragma unroll 1
  for (int k = 0; k < KNBR; ++k) {
    const int n = __shfl(nidx, k, 32);
    const float2 g = ((const float2*)(Qg + (size_t)n * 64))[lane];
    float p = g.x * q2v.x + g.y * q2v.y;
#pragma unroll
    for (int off = 16; off >= 1; off >>= 1) p += __shfl_xor(p, off, 32);
    if (lane == k) logit = p;
  }

  if (lane < KNBR) {
    logit = (logit + pe[(size_t)m * 32 + posi]) * 0.125f;   // 1/sqrt(64)
    if (!valid) logit = -10000.0f;
  } else {
    logit = -1e30f;
  }

  float mx = logit;
#pragma unroll
  for (int off = 16; off >= 1; off >>= 1) mx = fmaxf(mx, __shfl_xor(mx, off, 32));
  const float e = (lane < KNBR) ? __expf(logit - mx) : 0.0f;
  float s = e;
#pragma unroll
  for (int off = 16; off >= 1; off >>= 1) s += __shfl_xor(s, off, 32);
  const float aff = e / s;

  if (lane < KNBR) {
    out_aff[(size_t)m * KNBR + lane]   = aff;
    out_nidx[(size_t)m * KNBR + lane]  = (float)nidx;   // < 2^24: exact
    out_vmask[(size_t)m * KNBR + lane] = valid ? 1.0f : 0.0f;
  }

  // vsum = sem[m] + sum_k aff_k * sem[n_k]   (96 cols over 32 lanes x 3)
  float a0 = sem[(size_t)m * CSEM + lane];
  float a1 = sem[(size_t)m * CSEM + 32 + lane];
  float a2 = sem[(size_t)m * CSEM + 64 + lane];
#pragma unroll 1
  for (int k = 0; k < KNBR; ++k) {
    const int n = __shfl(nidx, k, 32);
    const float a = __shfl(aff, k, 32);
    const float* sp = sem + (size_t)n * CSEM;
    a0 += a * sp[lane];
    a1 += a * sp[32 + lane];
    a2 += a * sp[64 + lane];
  }
  vsum[(size_t)m * CSEM + lane]      = a0;
  vsum[(size_t)m * CSEM + 32 + lane] = a1;
  vsum[(size_t)m * CSEM + 64 + lane] = a2;
}

// ---------- Kernel 3: refined_feat = (vsum@W_v)@W_out + b_out; logits = @W_cls + b_cls ----------

__global__ __launch_bounds__(128) void k3_out(
    const float* __restrict__ vsum, const float* __restrict__ Wv,
    const float* __restrict__ Wout, const float* __restrict__ bout,
    const float* __restrict__ Wcls, const float* __restrict__ bcls,
    float* __restrict__ out_rf, float* __restrict__ out_logits)
{
  __shared__ __align__(16) _Float16 sWvT[64 * 104];  // N-major [n][k<96]
  __shared__ __align__(16) _Float16 sWoT[64 * 72];
  __shared__ __align__(16) _Float16 sWcT[16 * 72];
  __shared__ __align__(16) float    sF32[4][16 * 66];
  __shared__ __align__(16) _Float16 sF16[4][16 * 104];

  const int tid = threadIdx.x, lane = tid & 31, w = tid >> 5;

  for (int i = tid; i < 96 * 64; i += 128) {
    const int k = i >> 6, n = i & 63;
    sWvT[n * 104 + k] = (_Float16)Wv[i];
  }
  for (int i = tid; i < 64 * 64; i += 128) {
    const int k = i >> 6, n = i & 63;
    sWoT[n * 72 + k] = (_Float16)Wout[i];
  }
  for (int i = tid; i < 16 * 64; i += 128) {
    const int n = i >> 6, k = i & 63;
    sWcT[n * 72 + k] = (n < NCLS) ? (_Float16)Wcls[k * NCLS + n] : (_Float16)0.0f;
  }

  int t = blockIdx.x * 4 + w;
  if (t > MPTS / 16 - 1) t = MPTS / 16 - 1;
  const int m0 = t * 16;

  _Float16* a16 = sF16[w];
  float*    a32 = sF32[w];

  // stage vsum tile (16x96): float4 -> h4
  for (int idx = lane; idx < 16 * 24; idx += 32) {
    const int r = idx / 24, c4 = (idx % 24) << 2;
    const float4 g = *(const float4*)(vsum + (size_t)(m0 + r) * 96 + c4);
    h4 y; y[0] = (_Float16)g.x; y[1] = (_Float16)g.y;
    y[2] = (_Float16)g.z; y[3] = (_Float16)g.w;
    *(h4*)(a16 + r * 104 + c4) = y;
  }
  __syncthreads();

  // refined = vsum @ W_v (K=96 -> 3 k-steps)
  v8f acc[4];
#pragma unroll
  for (int nt = 0; nt < 4; ++nt) acc[nt] = zero8();
#pragma unroll
  for (int ks = 0; ks < 3; ++ks) {
    v16h a = load_a_frag(a16, 104, ks, lane);
#pragma unroll
    for (int nt = 0; nt < 4; ++nt)
      acc[nt] = wmma_f16(a, load_b_frag(sWvT, 104, nt, ks, lane), acc[nt]);
  }
#pragma unroll
  for (int nt = 0; nt < 4; ++nt) store_c_f32(a32, 66, nt, lane, acc[nt]);
  __syncthreads();
  for (int idx = lane; idx < 16 * 32; idx += 32) {
    const int r = idx >> 5, c2 = (idx & 31) << 1;
    const float2 x = *(const float2*)(a32 + r * 66 + c2);
    h2 y; y[0] = (_Float16)x.x; y[1] = (_Float16)x.y;
    *(h2*)(a16 + r * 104 + c2) = y;
  }
  __syncthreads();

  // refined_feat = refined @ W_out + b_out
#pragma unroll
  for (int nt = 0; nt < 4; ++nt) acc[nt] = zero8();
#pragma unroll
  for (int ks = 0; ks < 2; ++ks) {
    v16h a = load_a_frag(a16, 104, ks, lane);
#pragma unroll
    for (int nt = 0; nt < 4; ++nt)
      acc[nt] = wmma_f16(a, load_b_frag(sWoT, 72, nt, ks, lane), acc[nt]);
  }
  {
    const int rb = (lane >> 4) << 3;
#pragma unroll
    for (int nt = 0; nt < 4; ++nt) {
      const int col = (nt << 4) + (lane & 15);
      const float bo = bout[col];
#pragma unroll
      for (int r = 0; r < 8; ++r) {
        const float v = acc[nt][r] + bo;
        out_rf[(size_t)(m0 + rb + r) * 64 + col] = v;
        a32[(rb + r) * 66 + col] = v;
      }
    }
  }
  __syncthreads();
  for (int idx = lane; idx < 16 * 32; idx += 32) {
    const int r = idx >> 5, c2 = (idx & 31) << 1;
    const float2 x = *(const float2*)(a32 + r * 66 + c2);
    h2 y; y[0] = (_Float16)x.x; y[1] = (_Float16)x.y;
    *(h2*)(a16 + r * 104 + c2) = y;
  }
  __syncthreads();

  // logits = refined_feat @ W_cls (N padded 13->16) + b_cls
  {
    v8f c = zero8();
    c = wmma_f16(load_a_frag(a16, 104, 0, lane), load_b_frag(sWcT, 72, 0, 0, lane), c);
    c = wmma_f16(load_a_frag(a16, 104, 1, lane), load_b_frag(sWcT, 72, 0, 1, lane), c);
    const int rb = (lane >> 4) << 3, col = lane & 15;
    if (col < NCLS) {
      const float bc = bcls[col];
#pragma unroll
      for (int r = 0; r < 8; ++r)
        out_logits[(size_t)(m0 + rb + r) * NCLS + col] = c[r] + bc;
    }
  }
}

// ---------- launch ----------

extern "C" void kernel_launch(void* const* d_in, const int* in_sizes, int n_in,
                              void* d_out, int out_size, void* d_ws, size_t ws_size,
                              hipStream_t stream) {
  (void)in_sizes; (void)n_in; (void)out_size; (void)ws_size;

  const float*     geo     = (const float*)d_in[0];
  const float*     sem     = (const float*)d_in[1];
  const int*       coords  = (const int*)d_in[2];
  const long long* nbr     = (const long long*)d_in[3];
  const float*     Wgeo    = (const float*)d_in[4];
  const float*     gma     = (const float*)d_in[5];
  const float*     bta     = (const float*)d_in[6];
  const float*     Wbdy    = (const float*)d_in[7];
  const float*     bbdy    = (const float*)d_in[8];
  const float*     Wq      = (const float*)d_in[9];
  const float*     Wk      = (const float*)d_in[10];
  const float*     Wv      = (const float*)d_in[11];
  const float*     pos_emb = (const float*)d_in[12];
  const float*     Wout    = (const float*)d_in[13];
  const float*     bout    = (const float*)d_in[14];
  const float*     Wcls    = (const float*)d_in[15];
  const float*     bcls    = (const float*)d_in[16];

  float* out        = (float*)d_out;
  float* out_logits = out;                                   // [M,13]
  float* out_bdy    = out_logits + (size_t)MPTS * NCLS;      // [M,1]
  float* out_aff    = out_bdy + MPTS;                        // [M,1,27]
  float* out_rf     = out_aff + (size_t)MPTS * KNBR;         // [M,64]
  float* out_nidx   = out_rf + (size_t)MPTS * DDIM;          // [M,27]
  float* out_vmask  = out_nidx + (size_t)MPTS * KNBR;        // [M,27]

  float* ws = (float*)d_ws;
  float* Qg = ws;                                            // M*64
  float* q2 = Qg + (size_t)MPTS * 64;                        // M*64
  float* pe = q2 + (size_t)MPTS * 64;                        // M*32
  float* vs = pe + (size_t)MPTS * 32;                        // M*96

  const int tiles = MPTS / 16;          // 6250
  const int blk1  = (tiles + 3) / 4;    // 4 tiles (waves) per 128-thread block

  k1_geo<<<blk1, 128, 0, stream>>>(geo, Wgeo, gma, bta, Wbdy, bbdy, Wq, Wk,
                                   pos_emb, Qg, q2, pe, out_bdy);
  k2_attn<<<(MPTS + 3) / 4, 128, 0, stream>>>(Qg, q2, pe, sem, coords, nbr,
                                              out_aff, out_nidx, out_vmask, vs);
  k3_out<<<blk1, 128, 0, stream>>>(vs, Wv, Wout, bout, Wcls, bcls,
                                   out_rf, out_logits);
}
